// MultiHeadAttention_49572512530668
// MI455X (gfx1250) — matrix operand order repaired and checked by reading, not compiled, not run on hardware
//
#include <hip/hip_runtime.h>
#include <hip/hip_bf16.h>

// ---------------------------------------------------------------------------
// Multi-head attention forward for gfx1250 (MI455X), wave32 + WMMA bf16.
// Outputs (concatenated in d_out, float32): y, att, k, v
// Design: bf16 WMMA with f32 accumulation everywhere; att computed, softmaxed
// AND multiplied by V inside one kernel so the 537MB att tensor is written to
// HBM exactly once and never re-read (HBM-bound term: ~537MB @ 23.3TB/s).
// ---------------------------------------------------------------------------

#define DIM    1024
#define T_SEQ  2048
#define NH     16
#define HD     64
#define BATCH  2
#define BH     (BATCH * NH)   // 32
#define LDP    (T_SEQ + 4)    // padded LDS row stride (dwords): stripes banks

#define NEG_INF (-__builtin_huge_valf())

typedef __attribute__((ext_vector_type(16))) __bf16 v16bf;
typedef __attribute__((ext_vector_type(8)))  __bf16 v8bf;
typedef __attribute__((ext_vector_type(8)))  float  v8f;
typedef __attribute__((ext_vector_type(4)))  float  v4f;

__device__ __forceinline__ void lane_info(int& hf, int& l16) {
    int lane = threadIdx.x & 31;
    hf  = lane >> 4;   // 0: lanes 0-15, 1: lanes 16-31
    l16 = lane & 15;
}

__device__ __forceinline__ v8f bwmma(v16bf a, v16bf b, v8f c) {
    // D = A(16x32 bf16) x B(32x16 bf16) + C(16x16 f32)
    return __builtin_amdgcn_wmma_f32_16x16x32_bf16(
        /*neg_a=*/false, a, /*neg_b=*/false, b,
        /*c_mod=*/(short)0, c, /*reuse_a=*/false, /*reuse_b=*/false);
}

// A fragment (16x32 bf16) from f32 row-major, converting in-register.
// Lanes 0-15: M=lane, K {k0+0..7, k0+16..23}; lanes 16-31: K {+8..15, +24..31}.
__device__ __forceinline__ v16bf load_a_f32(const float* __restrict__ base,
                                            int ld, int m0, int k0) {
    int hf, m; lane_info(hf, m);
    const float* p = base + (size_t)(m0 + m) * ld + k0 + hf * 8;
    v8f x0 = *(const v8f*)p;
    v8f x1 = *(const v8f*)(p + 16);
    v16bf a;
#pragma unroll
    for (int e = 0; e < 8; ++e) { a[e] = (__bf16)x0[e]; a[e + 8] = (__bf16)x1[e]; }
    return a;
}

// A fragment from bf16 row-major.
__device__ __forceinline__ v16bf load_a_bf16(const __bf16* __restrict__ base,
                                             int ld, int m0, int k0) {
    int hf, m; lane_info(hf, m);
    const __bf16* p = base + (size_t)(m0 + m) * ld + k0 + hf * 8;
    union { v16bf v; v8bf h[2]; } u;
    u.h[0] = *(const v8bf*)p;
    u.h[1] = *(const v8bf*)(p + 16);
    return u.v;
}

// A fragment from the padded f32 probability strip in LDS.
__device__ __forceinline__ v16bf load_a_lds(const float* sS, int k0) {
    int hf, m; lane_info(hf, m);
    const float* p = sS + m * LDP + k0 + hf * 8;
    v4f x0 = *(const v4f*)p;
    v4f x1 = *(const v4f*)(p + 4);
    v4f x2 = *(const v4f*)(p + 16);
    v4f x3 = *(const v4f*)(p + 20);
    v16bf a;
#pragma unroll
    for (int e = 0; e < 4; ++e) {
        a[e]      = (__bf16)x0[e];
        a[e + 4]  = (__bf16)x1[e];
        a[e + 8]  = (__bf16)x2[e];
        a[e + 12] = (__bf16)x3[e];
    }
    return a;
}

// B fragment (32x16 bf16): memory holds Bᵀ row-major (N x K); lane n reads
// K {k0+half*8..+7, k0+16+half*8..+7} of column n (two 16B loads).
__device__ __forceinline__ v16bf load_b_bf16(const __bf16* __restrict__ base,
                                             int ld, int n0, int k0) {
    int hf, n; lane_info(hf, n);
    const __bf16* p = base + (size_t)(n0 + n) * ld + k0 + hf * 8;
    union { v16bf v; v8bf h[2]; } u;
    u.h[0] = *(const v8bf*)p;
    u.h[1] = *(const v8bf*)(p + 16);
    return u.v;
}

// ---------------------------------------------------------------------------
// K0: bulk f32 -> bf16 conversion (weights), pure bandwidth.
// ---------------------------------------------------------------------------
__global__ void cvt_bf16_kernel(const float* __restrict__ src,
                                __bf16* __restrict__ dst, int n8) {
    int i = (blockIdx.x * blockDim.x + threadIdx.x) * 8;
    if (i < n8 * 8) {
        v8f x = *(const v8f*)(src + i);
        v8bf y;
#pragma unroll
        for (int e = 0; e < 8; ++e) y[e] = (__bf16)x[e];
        *(v8bf*)(dst + i) = y;
    }
}

// ---------------------------------------------------------------------------
// K1: out = X @ W.T   (X: 4096x1024 f32, Wb: 1024x1024 bf16, N x K storage)
// Writes f32 split-head (b,h,t,hd) if outSplit != nullptr, and bf16 either
// split-head (transposeB16=0) or per-head transposed (b,h,hd,t) (=1).
// grid: (DIM/64, 4096/128), block: 128 (4 waves); wave -> 32(M) x 64(N) tile.
// ---------------------------------------------------------------------------
__global__ void proj_kernel(const float* __restrict__ X, const __bf16* __restrict__ Wb,
                            float* __restrict__ outSplit, __bf16* __restrict__ outB,
                            int transposeB16) {
    int wave = threadIdx.x >> 5;
    int m0 = blockIdx.y * 128 + wave * 32;
    int n0 = blockIdx.x * 64;

    v8f zero = {0.f, 0.f, 0.f, 0.f, 0.f, 0.f, 0.f, 0.f};
    v8f acc[2][4];
#pragma unroll
    for (int s = 0; s < 2; ++s)
#pragma unroll
        for (int t = 0; t < 4; ++t) acc[s][t] = zero;

    for (int k0 = 0; k0 < DIM; k0 += 32) {
        v16bf a0 = load_a_f32(X, DIM, m0, k0);
        v16bf a1 = load_a_f32(X, DIM, m0 + 16, k0);
#pragma unroll
        for (int t = 0; t < 4; ++t) {
            v16bf b = load_b_bf16(Wb, DIM, n0 + 16 * t, k0);
            acc[0][t] = bwmma(a0, b, acc[0][t]);
            acc[1][t] = bwmma(a1, b, acc[1][t]);
        }
    }

    int hf, nl; lane_info(hf, nl);
#pragma unroll
    for (int s = 0; s < 2; ++s) {
#pragma unroll
        for (int t = 0; t < 4; ++t) {
#pragma unroll
            for (int r = 0; r < 8; ++r) {
                int m = m0 + s * 16 + r + 8 * hf;  // global row (b*T + t)
                int n = n0 + 16 * t + nl;          // global col (h*HD + hd)
                int bb = m >> 11, tt = m & 2047;
                int h  = n >> 6,  hd = n & 63;
                size_t si = (((size_t)(bb * NH + h) * T_SEQ) + tt) * HD + hd;
                float val = acc[s][t][r];
                if (outSplit) outSplit[si] = val;
                if (!transposeB16)
                    outB[si] = (__bf16)val;
                else
                    outB[(((size_t)(bb * NH + h) * HD) + hd) * T_SEQ + tt] = (__bf16)val;
            }
        }
    }
}

// ---------------------------------------------------------------------------
// K2 (fused): causal scores + softmax + att@V for one (bh, 16-row strip).
// Phase 1: 8 waves fill the 16 x 2048 f32 score strip in padded LDS with WMMA
//          QKᵀ tiles up to the diagonal (causal skip = half the FLOPs).
// Phase 2: per-row wave32 softmax; normalized probs go to global att (the only
//          HBM write of att) AND back into LDS.
// Phase 3: y_heads strip = P @ V with A sourced from LDS, B = Vᵀ bf16 (L2).
//          8 waves = 4 hd-tiles x 2 K-halves; partials reduced via LDS.
// grid: (T/16, BH), block 256, dynShared: (16*LDP + 2*16*64)*4 = ~139.5 KB.
// ---------------------------------------------------------------------------
__global__ void attn_fused_kernel(const __bf16* __restrict__ qb,
                                  const __bf16* __restrict__ kb,
                                  const __bf16* __restrict__ vtb,
                                  float* __restrict__ att,
                                  __bf16* __restrict__ yhb) {
    extern __shared__ float sS[];                 // [16][LDP] + psum[2][16][64]
    float* psum = sS + 16 * LDP;

    int bh = blockIdx.y;
    int i0 = blockIdx.x * 16;
    const __bf16* qh = qb  + (size_t)bh * T_SEQ * HD;
    const __bf16* kh = kb  + (size_t)bh * T_SEQ * HD;
    const __bf16* vt = vtb + (size_t)bh * HD * T_SEQ;   // (hd, t)

    int wave = threadIdx.x >> 5;
    int lane = threadIdx.x & 31;
    int hf, nl; lane_info(hf, nl);

    const float scale = 0.03125f;                 // 1/sqrt(1024)
    int nTiles = (i0 >> 4) + 1;                   // j-tiles up to + incl. diagonal
    v8f zero = {0.f, 0.f, 0.f, 0.f, 0.f, 0.f, 0.f, 0.f};

    // ---- Phase 1: scores into LDS -------------------------------------
    v16bf a0 = load_a_bf16(qh, HD, i0, 0);
    v16bf a1 = load_a_bf16(qh, HD, i0, 32);

    for (int jt = wave; jt < nTiles; jt += 8) {   // wave-uniform loop
        int j0 = jt * 16;
        v16bf b0 = load_b_bf16(kh, HD, j0, 0);
        v16bf b1 = load_b_bf16(kh, HD, j0, 32);
        v8f acc = zero;
        acc = bwmma(a0, b0, acc);
        acc = bwmma(a1, b1, acc);
#pragma unroll
        for (int r = 0; r < 8; ++r) {
            int rl = r + 8 * hf;
            int i  = i0 + rl;
            int j  = j0 + nl;
            sS[rl * LDP + j] = (j <= i) ? acc[r] * scale : NEG_INF;
        }
    }
    __syncthreads();

    // ---- Phase 2: softmax; write att once; probs back to LDS ----------
#pragma unroll
    for (int rr = 0; rr < 2; ++rr) {              // 8 waves x 2 rows = 16 rows
        int row = wave * 2 + rr;
        int i   = i0 + row;
        float* sr = sS + row * LDP;

        float mx = NEG_INF;
        for (int j = lane; j <= i; j += 32) mx = fmaxf(mx, sr[j]);
        for (int off = 16; off > 0; off >>= 1) mx = fmaxf(mx, __shfl_xor(mx, off, 32));

        float sum = 0.f;
        for (int j = lane; j <= i; j += 32) sum += __expf(sr[j] - mx);
        for (int off = 16; off > 0; off >>= 1) sum += __shfl_xor(sum, off, 32);
        float inv = 1.0f / sum;

        float* orow = att + ((size_t)bh * T_SEQ + i) * T_SEQ;
        for (int j = lane; j < T_SEQ; j += 32) {
            float pv = (j <= i) ? __expf(sr[j] - mx) * inv : 0.0f;
            orow[j] = pv;
            sr[j]   = pv;
        }
    }
    __syncthreads();

    // ---- Phase 3: y_heads strip = P @ V from LDS ----------------------
    {
        int nt  = wave & 3;                       // hd tile (0..3)
        int kh2 = wave >> 2;                      // K half (0..1)
        int n0  = nt * 16;
        v8f acc = zero;
        int kmax = i0 + 16;                       // probs are zero past diagonal
        for (int k0 = kh2 * 32; k0 < kmax; k0 += 64) {
            v16bf a = load_a_lds(sS, k0);
            v16bf b = load_b_bf16(vt, T_SEQ, n0, k0);
            acc = bwmma(a, b, acc);
        }
        float* ps = psum + kh2 * (16 * HD);
#pragma unroll
        for (int r = 0; r < 8; ++r)
            ps[(r + 8 * hf) * HD + n0 + nl] = acc[r];
    }
    __syncthreads();

    for (int e = threadIdx.x; e < 16 * HD; e += 256) {
        int rl = e >> 6, hd = e & 63;
        float v = psum[e] + psum[16 * HD + e];
        yhb[((size_t)bh * T_SEQ + i0 + rl) * HD + hd] = (__bf16)v;
    }
}

// A fragment from head-split bf16 y_heads (b,h,t,hd) viewed as merged (m,1024).
// Each 8-element K-chunk stays inside one 64-wide head, so loads are contiguous.
__device__ __forceinline__ v16bf load_a_yh(const __bf16* __restrict__ yhb,
                                           int m0, int k0) {
    int hf, m; lane_info(hf, m);
    int mm = m0 + m;
    int bb = mm >> 11, tt = mm & 2047;
    int s0 = k0 + hf * 8;
    int s1 = s0 + 16;
    const __bf16* p0 = yhb + (((size_t)(bb * NH + (s0 >> 6)) * T_SEQ + tt) * HD) + (s0 & 63);
    const __bf16* p1 = yhb + (((size_t)(bb * NH + (s1 >> 6)) * T_SEQ + tt) * HD) + (s1 & 63);
    union { v16bf v; v8bf h[2]; } u;
    u.h[0] = *(const v8bf*)p0;
    u.h[1] = *(const v8bf*)p1;
    return u.v;
}

// ---------------------------------------------------------------------------
// K3: y = merge_heads(y_heads) @ Wp.T  -> f32 row-major (b,t,d).
// grid: (DIM/64, 4096/128), block 128 (4 waves); wave -> 32 x 64 tile.
// ---------------------------------------------------------------------------
__global__ void outproj_kernel(const __bf16* __restrict__ yhb,
                               const __bf16* __restrict__ Wpb,
                               float* __restrict__ y) {
    int wave = threadIdx.x >> 5;
    int m0 = blockIdx.y * 128 + wave * 32;
    int n0 = blockIdx.x * 64;

    v8f zero = {0.f, 0.f, 0.f, 0.f, 0.f, 0.f, 0.f, 0.f};
    v8f acc[2][4];
#pragma unroll
    for (int s = 0; s < 2; ++s)
#pragma unroll
        for (int t = 0; t < 4; ++t) acc[s][t] = zero;

    for (int k0 = 0; k0 < DIM; k0 += 32) {
        v16bf a0 = load_a_yh(yhb, m0, k0);
        v16bf a1 = load_a_yh(yhb, m0 + 16, k0);
#pragma unroll
        for (int t = 0; t < 4; ++t) {
            v16bf b = load_b_bf16(Wpb, DIM, n0 + 16 * t, k0);
            acc[0][t] = bwmma(a0, b, acc[0][t]);
            acc[1][t] = bwmma(a1, b, acc[1][t]);
        }
    }

    int hf, nl; lane_info(hf, nl);
#pragma unroll
    for (int s = 0; s < 2; ++s)
#pragma unroll
        for (int t = 0; t < 4; ++t)
#pragma unroll
            for (int r = 0; r < 8; ++r)
                y[(size_t)(m0 + s * 16 + r + 8 * hf) * DIM + n0 + 16 * t + nl]
                    = acc[s][t][r];
}

// ---------------------------------------------------------------------------
extern "C" void kernel_launch(void* const* d_in, const int* in_sizes, int n_in,
                              void* d_out, int out_size, void* d_ws, size_t ws_size,
                              hipStream_t stream) {
    (void)in_sizes; (void)n_in; (void)out_size; (void)ws_size;

    const float* Q  = (const float*)d_in[0];
    const float* K  = (const float*)d_in[1];
    const float* V  = (const float*)d_in[2];
    const float* Wq = (const float*)d_in[3];
    const float* Wk = (const float*)d_in[4];
    const float* Wv = (const float*)d_in[5];
    const float* Wp = (const float*)d_in[6];

    // d_out layout: y | att | k | v  (all f32)
    float* y   = (float*)d_out;
    float* att = y   + (size_t)BATCH * T_SEQ * DIM;      // 4,194,304
    float* kO  = att + (size_t)BH * T_SEQ * T_SEQ;       // +134,217,728
    float* vO  = kO  + (size_t)BH * T_SEQ * HD;          // +4,194,304

    // workspace (bf16): q, k (split-head), vᵀ (b,h,hd,t), y_heads, 4x weights
    const size_t QKV = (size_t)BH * T_SEQ * HD;          // 4M elements
    __bf16* qb  = (__bf16*)d_ws;
    __bf16* kb  = qb  + QKV;
    __bf16* vtb = kb  + QKV;
    __bf16* yhb = vtb + QKV;
    __bf16* Wqb = yhb + QKV;
    __bf16* Wkb = Wqb + (size_t)DIM * DIM;
    __bf16* Wvb = Wkb + (size_t)DIM * DIM;
    __bf16* Wpb = Wvb + (size_t)DIM * DIM;               // total 40 MB

    // K0: weight conversion (1M elems each -> 512 blocks of 256)
    const int nW8 = (DIM * DIM) / 8;
    cvt_bf16_kernel<<<nW8 / 256, 256, 0, stream>>>(Wq, Wqb, nW8);
    cvt_bf16_kernel<<<nW8 / 256, 256, 0, stream>>>(Wk, Wkb, nW8);
    cvt_bf16_kernel<<<nW8 / 256, 256, 0, stream>>>(Wv, Wvb, nW8);
    cvt_bf16_kernel<<<nW8 / 256, 256, 0, stream>>>(Wp, Wpb, nW8);

    dim3 gProj(DIM / 64, (BATCH * T_SEQ) / 128);         // (16, 32)
    proj_kernel<<<gProj, 128, 0, stream>>>(Q, Wqb, nullptr, qb, 0);
    proj_kernel<<<gProj, 128, 0, stream>>>(K, Wkb, kO, kb, 0);
    proj_kernel<<<gProj, 128, 0, stream>>>(V, Wvb, vO, vtb, 1);

    const size_t ldsBytes = (16 * LDP + 2 * 16 * HD) * sizeof(float);
    attn_fused_kernel<<<dim3(T_SEQ / 16, BH), 256, ldsBytes, stream>>>(
        qb, kb, vtb, att, yhb);

    outproj_kernel<<<gProj, 128, 0, stream>>>(yhb, Wpb, y);
}
